// FastKANLayer_37056977830528
// MI455X (gfx1250) — matrix operand order repaired and checked by reading, not compiled, not run on hardware
//
#include <hip/hip_runtime.h>

#define DDIM 512
#define GRIDS 15
#define KPACK 16                  // 15 RBF slots + 1 relu slot per input dim
#define KTOT (DDIM * KPACK)       // 8192
#define MT 128
#define NT 128
#define KB 128                    // K-block: 8 input dims * 16 slots
#define NKB (KTOT / KB)           // 64
#define PITCH 136                 // LDS row pitch in bf16 elems (272 B: 16B aligned, bank-spread)
#define TILE_ELEMS (MT * PITCH)

typedef __attribute__((ext_vector_type(16))) __bf16 v16bf;
typedef __attribute__((ext_vector_type(8)))  float  v8f;
typedef unsigned int u32x4 __attribute__((ext_vector_type(4)));
typedef int          i32x4 __attribute__((ext_vector_type(4)));
typedef int          i32x8 __attribute__((ext_vector_type(8)));

union FragU { uint4 u[2]; v16bf v; };

#if defined(__HIP_DEVICE_COMPILE__) && __has_builtin(__builtin_amdgcn_tensor_load_to_lds)
#define USE_TDM 1
#else
#define USE_TDM 0
#endif

#if defined(__HIP_DEVICE_COMPILE__) && __has_builtin(__builtin_amdgcn_cvt_pk_bf16_f32)
#define HAVE_PK_BF16 1
#else
#define HAVE_PK_BF16 0
#endif

__device__ __forceinline__ unsigned short f2bf(float f) {
  unsigned int u = __float_as_uint(f);
  return (unsigned short)((u + 0x7FFFu + ((u >> 16) & 1u)) >> 16);  // RNE
}

__device__ __forceinline__ unsigned int pack_bf16(float lo, float hi) {
#if HAVE_PK_BF16
  auto r = __builtin_amdgcn_cvt_pk_bf16_f32(lo, hi);   // v_cvt_pk_bf16_f32
  unsigned int u;
  __builtin_memcpy(&u, &r, 4);
  return u;
#else
  return (unsigned int)f2bf(lo) | ((unsigned int)f2bf(hi) << 16);
#endif
}

// ---------------- weight prep: interleave spline + base into bf16 ----------------
// Wb[o][k], k = d*16+g:  g<15 -> spline_w[o][d*15+g] ; g==15 -> base_w[d][o]
__global__ void prep_weights(const float* __restrict__ spline_w,
                             const float* __restrict__ base_w,
                             unsigned short* __restrict__ Wb) {
  int idx = blockIdx.x * blockDim.x + threadIdx.x;   // 512*8192 exact
  int o = idx >> 13;
  int k = idx & (KTOT - 1);
  int d = k >> 4;
  int g = k & 15;
  float v = (g < GRIDS) ? spline_w[(size_t)o * (DDIM * GRIDS) + d * GRIDS + g]
                        : base_w[(size_t)d * DDIM + o];
  Wb[idx] = f2bf(v);
}

#if USE_TDM
// One TDM descriptor: 2D tile 128 rows x 128 bf16 from row-major W[512][8192],
// LDS-padded to 272 B row pitch (pad 16 B after every 256 B tile row).
__device__ __forceinline__ void tdm_load_B(const unsigned short* gsrc, unsigned int lds_off) {
  unsigned long long ga = (unsigned long long)(const void*)gsrc;
  u32x4 g0;
  g0[0] = 1u;                                       // count=1, user descriptor
  g0[1] = lds_off;                                  // lds_addr (bytes)
  g0[2] = (unsigned int)ga;                         // global_addr[31:0]
  g0[3] = (unsigned int)(ga >> 32) | (2u << 30);    // global_addr[56:32] | type=2
  i32x8 g1;
  g1[0] = (1 << 16)        // data_size = 2 bytes
        | (1 << 20)        // pad_enable
        | (5 << 22)        // pad_interval: every 64 DWORDs (= 256 B tile row)
        | (3 << 25);       // pad_amount: 4 DWORDs (16 B) -> 272 B LDS pitch
  g1[1] = (int)((KTOT & 0xFFFF) << 16);             // tensor_dim0[15:0] = 8192
  g1[2] = (int)((KTOT >> 16) | (DDIM << 16));       // tensor_dim0 hi | tensor_dim1[15:0]=512
  g1[3] = (int)(KB << 16);                          // tensor_dim1 hi | tile_dim0=128
  g1[4] = MT;                                       // tile_dim1=128 | tile_dim2=0
  g1[5] = KTOT;                                     // tensor_dim0_stride[31:0] = 8192
  g1[6] = 0;
  g1[7] = 0;
  i32x4 z4 = {0, 0, 0, 0};
#if __clang_major__ >= 23
  i32x8 z8 = {0, 0, 0, 0, 0, 0, 0, 0};
  __builtin_amdgcn_tensor_load_to_lds(g0, g1, z4, z4, z8, 0);
#else
  __builtin_amdgcn_tensor_load_to_lds(g0, g1, z4, z4, 0);
#endif
}
#endif

// ---------------- A-tile generation: LN + RBF basis + relu slot ----------------
// Each thread: one row, 4 input dims (float4 of x), 16 slots per dim.
__device__ __forceinline__ void gen_A(unsigned short* __restrict__ Abuf,
                                      const float* __restrict__ x,
                                      const float* __restrict__ gamma,
                                      const float* __restrict__ beta,
                                      const float2* __restrict__ stats,
                                      int mbase, int kb, int t) {
  const int n   = t >> 1;
  const int dd4 = (t & 1) << 2;                     // 0 or 4
  const int d0  = kb * 8;
  const float4 xv = *(const float4*)(x + (size_t)(mbase + n) * DDIM + d0 + dd4);
  const float xr4[4] = {xv.x, xv.y, xv.z, xv.w};
  const float mean = stats[n].x;
  const float rstd = stats[n].y;
  const float inv_denom = 14.0f / 4.0f;             // 1/DENOM = 3.5
  #pragma unroll
  for (int j = 0; j < 4; ++j) {
    const int d = d0 + dd4 + j;
    const float xraw = xr4[j];
    const float xn = (xraw - mean) * rstd * gamma[d] + beta[d];
    unsigned int pk[8];
    #pragma unroll
    for (int gp = 0; gp < 8; ++gp) {
      const int g0 = gp * 2, g1 = gp * 2 + 1;
      const float c0 = -2.0f + (float)g0 * (4.0f / 14.0f);
      const float u0 = (xn - c0) * inv_denom;
      const float lo = __expf(-u0 * u0);
      float hi;
      if (g1 < GRIDS) {
        const float c1 = -2.0f + (float)g1 * (4.0f / 14.0f);
        const float u1 = (xn - c1) * inv_denom;
        hi = __expf(-u1 * u1);
      } else {
        hi = fmaxf(xraw, 0.f);                      // relu(x) residual slot
      }
      pk[gp] = pack_bf16(lo, hi);
    }
    uint4* dst = (uint4*)&Abuf[n * PITCH + (dd4 + j) * 16];
    dst[0] = make_uint4(pk[0], pk[1], pk[2], pk[3]);
    dst[1] = make_uint4(pk[4], pk[5], pk[6], pk[7]);
  }
}

#if !USE_TDM
__device__ __forceinline__ void copy_B(unsigned short* __restrict__ Bbuf,
                                       const unsigned short* __restrict__ Wb,
                                       int nbase, int kb, int t) {
  #pragma unroll
  for (int i = 0; i < 8; ++i) {
    const int lin = i * 256 + t;
    const int oo  = lin >> 4;
    const int kkb = lin & 15;
    const uint4 v = *(const uint4*)(Wb + (size_t)(nbase + oo) * KTOT + kb * KB + kkb * 8);
    *(uint4*)&Bbuf[oo * PITCH + kkb * 8] = v;
  }
}
#endif

__global__ void __launch_bounds__(256, 1) kan_fused(
    const float* __restrict__ x,
    const float* __restrict__ gamma,
    const float* __restrict__ beta,
    const float* __restrict__ base_b,
    const unsigned short* __restrict__ Wb,
    float* __restrict__ out) {
  __shared__ __align__(16) unsigned short As[2][TILE_ELEMS];   // 2 x 34816 B
  __shared__ __align__(16) unsigned short Bs[2][TILE_ELEMS];   // 2 x 34816 B
  __shared__ float2 stats[MT];
  __shared__ float red[512];

  const int t = threadIdx.x;
  const int mbase = blockIdx.x * MT;
  const int nbase = blockIdx.y * NT;
  const int w = t >> 5;

  // ---------- LayerNorm statistics for the 128-row block ----------
  {
    const int row = t >> 1;
    const int half = t & 1;
    const float* xr = x + (size_t)(mbase + row) * DDIM + half * 256;
    float s = 0.f, ss = 0.f;
    #pragma unroll 4
    for (int i = 0; i < 256; i += 4) {
      float4 v = *(const float4*)(xr + i);
      s  += v.x + v.y + v.z + v.w;
      ss += v.x * v.x + v.y * v.y + v.z * v.z + v.w * v.w;
    }
    red[t] = s;
    red[256 + t] = ss;
  }
  __syncthreads();
  if ((t & 1) == 0) {
    const int row = t >> 1;
    float sum  = red[t] + red[t + 1];
    float sq   = red[256 + t] + red[257 + t];
    float mean = sum * (1.f / DDIM);
    float var  = sq * (1.f / DDIM) - mean * mean;
    stats[row] = make_float2(mean, rsqrtf(var + 1e-3f));
  }
  __syncthreads();

  // ---------- prolog: stage K-block 0 into buffer 0 ----------
#if USE_TDM
  if (w == 0)
    tdm_load_B(Wb + (size_t)nbase * KTOT,
               (unsigned int)(unsigned long long)(const void*)&Bs[0][0]);
#else
  copy_B(&Bs[0][0], Wb, nbase, 0, t);
#endif
  gen_A(&As[0][0], x, gamma, beta, stats, mbase, 0, t);
#if USE_TDM
  __builtin_amdgcn_s_wait_tensorcnt(0);   // no-op for waves that issued nothing
#endif
  __syncthreads();

  v8f acc[2][4] = {};

  const int wm   = (w & 3) * 32;      // wave M offset in tile
  const int wn   = (w >> 2) * 64;     // wave N offset in tile
  const int lane = t & 31;
  const int lr   = lane & 15;
  const int lhi  = lane >> 4;

  for (int kb = 0; kb < NKB; ++kb) {
    const unsigned short* Acur = &As[kb & 1][0];
    const unsigned short* Bcur = &Bs[kb & 1][0];
    // ---------- stage next K-block into the ping buffer ----------
    if (kb + 1 < NKB) {
#if USE_TDM
      if (w == 0)
        tdm_load_B(Wb + (size_t)nbase * KTOT + (size_t)(kb + 1) * KB,
                   (unsigned int)(unsigned long long)(const void*)&Bs[(kb + 1) & 1][0]);
#else
      copy_B(&Bs[(kb + 1) & 1][0], Wb, nbase, kb + 1, t);
#endif
      gen_A(&As[(kb + 1) & 1][0], x, gamma, beta, stats, mbase, kb + 1, t);
    }
    // ---------- WMMA compute on the pong buffer: 4 k-steps of 32, 2x4 tiles ----------
    #pragma unroll
    for (int ks = 0; ks < 4; ++ks) {
      const int kk = ks * 32;
      FragU af[2];
      #pragma unroll
      for (int i = 0; i < 2; ++i) {
        // A lane layout: row M=L%16; 16 contiguous K at (L/16)*16
        const uint4* p = (const uint4*)&Acur[(wm + i * 16 + lr) * PITCH + kk + lhi * 16];
        af[i].u[0] = p[0];
        af[i].u[1] = p[1];
      }
      FragU bfr[4];
      #pragma unroll
      for (int jn = 0; jn < 4; ++jn) {
        // B lane layout: col N=L%16; 16 contiguous K at (L/16)*16
        const uint4* p = (const uint4*)&Bcur[(wn + jn * 16 + lr) * PITCH + kk + lhi * 16];
        bfr[jn].u[0] = p[0];
        bfr[jn].u[1] = p[1];
      }
      #pragma unroll
      for (int i = 0; i < 2; ++i)
        #pragma unroll
        for (int jn = 0; jn < 4; ++jn)
          acc[i][jn] = __builtin_amdgcn_wmma_f32_16x16x32_bf16(
              false, af[i].v, false, bfr[jn].v, (short)0, acc[i][jn], false, false);
    }
#if USE_TDM
    __builtin_amdgcn_s_wait_tensorcnt(0);
#endif
    __syncthreads();
  }

  // ---------- epilogue: add bias, store ----------
  #pragma unroll
  for (int jn = 0; jn < 4; ++jn) {
    const int col = nbase + wn + jn * 16 + lr;
    const float bb = base_b[col];
    #pragma unroll
    for (int i = 0; i < 2; ++i) {
      const int rbase = mbase + wm + i * 16 + lhi * 8;   // VGPR v -> row v + 8*(lane/16)
      #pragma unroll
      for (int v = 0; v < 8; ++v) {
        out[(size_t)(rbase + v) * DDIM + col] = acc[i][jn][v] + bb;
      }
    }
  }
}

extern "C" void kernel_launch(void* const* d_in, const int* in_sizes, int n_in,
                              void* d_out, int out_size, void* d_ws, size_t ws_size,
                              hipStream_t stream) {
  const float* x        = (const float*)d_in[0];
  const float* gamma    = (const float*)d_in[1];
  const float* beta     = (const float*)d_in[2];
  const float* spline_w = (const float*)d_in[3];
  const float* base_w   = (const float*)d_in[4];
  const float* base_b   = (const float*)d_in[5];
  float* out = (float*)d_out;
  unsigned short* Wb = (unsigned short*)d_ws;   // 512*8192 bf16 = 8 MB

  const int N = in_sizes[0] / DDIM;             // 32768

  prep_weights<<<(DDIM * KTOT) / 256, 256, 0, stream>>>(spline_w, base_w, Wb);

  dim3 g(N / MT, DDIM / NT);                    // 256 x 4 workgroups
  kan_fused<<<g, 256, 0, stream>>>(x, gamma, beta, base_b, Wb, out);
}
